// TransformerLayer_17351667876365
// MI455X (gfx1250) — compile-verified, matrix operand order checked by reading, not run on hardware
//
#include <hip/hip_runtime.h>
#include <math.h>

// ---------------- CDNA5 WMMA types ----------------
typedef __bf16  v16bf __attribute__((ext_vector_type(16)));
typedef float   v8f   __attribute__((ext_vector_type(8)));

static __device__ __forceinline__ v8f wmma_bf16(v16bf a, v16bf b, v8f c) {
  // D = A(16x32 bf16) x B(32x16 bf16) + C(16x16 f32)
  return __builtin_amdgcn_wmma_f32_16x16x32_bf16(false, a, false, b, (short)0, c, false, false);
}

// Fragment loaders per CDNA5 ISA 16-bit 16x32 layout:
// lanes 0-15 hold K = [0..7, 16..23], lanes 16-31 hold K = [8..15, 24..31]
// caller passes c0 = 8*(lane>>4). bf16 rows -> two contiguous 16B loads.
static __device__ __forceinline__ v16bf frag_bf16_row(const __bf16* __restrict__ p, int c0) {
  v16bf f;
#pragma unroll
  for (int j = 0; j < 8; ++j) {
    f[j]     = p[c0 + j];
    f[j + 8] = p[c0 + 16 + j];
  }
  return f;
}

static __device__ __forceinline__ float gelu_tanh(float x) {
  const float k0 = 0.7978845608028654f, k1 = 0.044715f;
  float t = tanhf(k0 * (x + k1 * x * x * x));
  return 0.5f * x * (1.0f + t);
}

// ---------------- problem constants ----------------
#define SEQ 2048
#define BATCH 2
#define HID 1024
#define NHEADS 16
#define NKV 4
#define HDIM 64
#define NEXP 8
#define FFN 4096
#define TOKENS (SEQ * BATCH)              // 4096
#define QKVW ((NHEADS + 2 * NKV) * HDIM)  // 1536
#define MAX_T32 264                       // ceil((2*T + 8*31)/32), 8-divisible
#define MAX_ROWS (MAX_T32 * 32)           // 8448

// ---------------- LayerNorm (optional f32 and bf16 outputs) ----------------
__global__ __launch_bounds__(256) void ln_kernel(const float* __restrict__ x,
                                                 const float* __restrict__ w,
                                                 const float* __restrict__ b,
                                                 float* __restrict__ yf,
                                                 __bf16* __restrict__ yb) {
  const int H = HID;
  int t = blockIdx.x, tid = threadIdx.x;
  const float* xr = x + (size_t)t * H;
  float v[4], s = 0.f;
#pragma unroll
  for (int j = 0; j < 4; ++j) { v[j] = xr[tid + j * 256]; s += v[j]; }
  __shared__ float red[256];
  red[tid] = s; __syncthreads();
  for (int o = 128; o > 0; o >>= 1) { if (tid < o) red[tid] += red[tid + o]; __syncthreads(); }
  float mu = red[0] * (1.f / H);
  __syncthreads();
  s = 0.f;
#pragma unroll
  for (int j = 0; j < 4; ++j) { float d = v[j] - mu; s += d * d; }
  red[tid] = s; __syncthreads();
  for (int o = 128; o > 0; o >>= 1) { if (tid < o) red[tid] += red[tid + o]; __syncthreads(); }
  float rstd = rsqrtf(red[0] * (1.f / H) + 1e-5f);
#pragma unroll
  for (int j = 0; j < 4; ++j) {
    int i = tid + j * 256;
    float o = (v[j] - mu) * rstd * w[i] + b[i];
    if (yf) yf[(size_t)t * H + i] = o;
    if (yb) yb[(size_t)t * H + i] = (__bf16)o;
  }
}

// ---------------- one-shot f32 -> bf16 convert ----------------
__global__ __launch_bounds__(256) void cvt_bf16_kernel(const float* __restrict__ src,
                                                       __bf16* __restrict__ dst, size_t n) {
  size_t i0 = ((size_t)blockIdx.x * 256 + threadIdx.x) * 4;
#pragma unroll
  for (int j = 0; j < 4; ++j) {
    size_t i = i0 + j;
    if (i < n) dst[i] = (__bf16)src[i];
  }
}

// ---------------- one-shot weight transpose: [E][K][N] f32 -> [E][N][K] bf16 ----------------
__global__ __launch_bounds__(256) void transpose_w_kernel(const float* __restrict__ src,
                                                          __bf16* __restrict__ dst,
                                                          int K, int N) {
  __shared__ __bf16 tile[64][65];
  int e = blockIdx.z;
  int kb = blockIdx.x * 64, nb = blockIdx.y * 64;
  const float* s = src + (size_t)e * K * N;
  __bf16* d = dst + (size_t)e * N * K;
  int tx = threadIdx.x & 63, ty = threadIdx.x >> 6;  // 4 row-groups
#pragma unroll
  for (int j = 0; j < 16; ++j) {
    int r = ty + j * 4;
    tile[r][tx] = (__bf16)s[(size_t)(kb + r) * N + (nb + tx)];
  }
  __syncthreads();
#pragma unroll
  for (int j = 0; j < 16; ++j) {
    int r = ty + j * 4;
    d[(size_t)(nb + r) * K + (kb + tx)] = tile[tx][r];
  }
}

// ---------------- unified bf16 WMMA GEMM ----------------
// C[M,N] = op( A[M,K](row, optional gather) x W[N,K]^T ) (+Res), per-expert W offset.
// One wave computes a 32x64 tile: 2 A-frags x 4 B-frags -> 8 WMMAs per k-step.
template <int GATHER, int GELU, int OUTBF, int RES>
__global__ __launch_bounds__(256) void gemm_bf(const __bf16* __restrict__ A,
                                               const __bf16* __restrict__ W,
                                               const int* __restrict__ perm,
                                               const int* __restrict__ tileExp,
                                               const float* __restrict__ Res,
                                               void* __restrict__ Cout,
                                               int M, int N, int K, long wStride) {
  int lane = threadIdx.x & 31, wv = threadIdx.x >> 5;
  int wt = blockIdx.x * 8 + wv;       // 32-row wave tile
  int m0 = wt * 32;
  if (m0 >= M) return;
  long woff = 0;
  if (tileExp) { int e = tileExp[wt]; if (e < 0) return; woff = (long)e * wStride; }
  int n0 = blockIdx.y * 64;
  int lr = lane & 15, half = lane >> 4, c0 = half * 8;
  const __bf16 *a0, *a1;
  if (GATHER) {
    int t0 = perm[m0 + lr];      if (t0 < 0) t0 = 0;
    int t1 = perm[m0 + 16 + lr]; if (t1 < 0) t1 = 0;
    a0 = A + (size_t)t0 * K;
    a1 = A + (size_t)t1 * K;
  } else {
    a0 = A + (size_t)(m0 + lr) * K;
    a1 = a0 + (size_t)16 * K;
  }
  const __bf16* wb = W + woff;
  v8f acc[2][4] = {};
  for (int k0 = 0; k0 < K; k0 += 32) {
    v16bf af0 = frag_bf16_row(a0 + k0, c0);
    v16bf af1 = frag_bf16_row(a1 + k0, c0);
#pragma unroll
    for (int s = 0; s < 4; ++s) {
      v16bf bf = frag_bf16_row(wb + (size_t)(n0 + s * 16 + lr) * K + k0, c0);
      acc[0][s] = wmma_bf16(af0, bf, acc[0][s]);
      acc[1][s] = wmma_bf16(af1, bf, acc[1][s]);
    }
  }
#pragma unroll
  for (int mi = 0; mi < 2; ++mi)
#pragma unroll
    for (int s = 0; s < 4; ++s)
#pragma unroll
      for (int r = 0; r < 8; ++r) {
        size_t row = (size_t)(m0 + mi * 16 + r + 8 * half);
        size_t col = (size_t)(n0 + s * 16 + lr);
        float v = acc[mi][s][r];
        if (GELU) v = gelu_tanh(v);
        if (RES) v += Res[row * N + col];
        if (OUTBF) ((__bf16*)Cout)[row * N + col] = (__bf16)v;
        else       ((float*)Cout)[row * N + col] = v;
      }
}

// ---------------- flash attention (causal, GQA), bf16 QKV ----------------
// one wave per 16-query block per (batch, head); qkv = [T, 1536] bf16
__global__ __launch_bounds__(256) void attn_kernel(const __bf16* __restrict__ qkv,
                                                   __bf16* __restrict__ out) {
  int lane = threadIdx.x & 31, wv = threadIdx.x >> 5;
  int gw = blockIdx.x * 8 + wv;
  int qb = gw & 127;
  int bb = (gw >> 7) & 1;
  int h  = gw >> 8;  // 0..15
  int s0 = qb * 16;
  int lr = lane & 15, half = lane >> 4, c0 = half * 8;
  const size_t TS = (size_t)BATCH * QKVW;

  const __bf16* qptr = qkv + ((size_t)(s0 + lr) * BATCH + bb) * QKVW + h * HDIM;
  v16bf aq[2];
#pragma unroll
  for (int c = 0; c < 2; ++c) aq[c] = frag_bf16_row(qptr + 32 * c, c0);

  const __bf16* kbase = qkv + (size_t)bb * QKVW + NHEADS * HDIM + (h >> 2) * HDIM;
  const __bf16* vbase = kbase + NKV * HDIM;

  const float NEG = -__builtin_inff();
  float m_i[8], l_i[8], alpha[8];
#pragma unroll
  for (int r = 0; r < 8; ++r) { m_i[r] = NEG; l_i[r] = 0.f; }
  v8f o[4] = {};

  __shared__ __bf16 pbuf[8][512];
  __bf16* pb = pbuf[wv];

  int ktmax = (s0 + 15) >> 5;
  for (int kt = 0; kt <= ktmax; ++kt) {
    v8f sc[2] = {};
#pragma unroll
    for (int sub = 0; sub < 2; ++sub) {
      const __bf16* kr = kbase + (size_t)(kt * 32 + sub * 16 + lr) * TS;
#pragma unroll
      for (int c = 0; c < 2; ++c) {
        v16bf bf = frag_bf16_row(kr + 32 * c, c0);
        sc[sub] = wmma_bf16(aq[c], bf, sc[sub]);
      }
    }
#pragma unroll
    for (int sub = 0; sub < 2; ++sub)
#pragma unroll
      for (int r = 0; r < 8; ++r) {
        int key = kt * 32 + sub * 16 + lr;
        int qrow = s0 + r + 8 * half;
        float v = sc[sub][r] * 0.125f;
        sc[sub][r] = (key > qrow) ? NEG : v;
      }
#pragma unroll
    for (int r = 0; r < 8; ++r) {
      float v = fmaxf(sc[0][r], sc[1][r]);
#pragma unroll
      for (int off = 1; off < 16; off <<= 1) v = fmaxf(v, __shfl_xor(v, off, 32));
      float mn = fmaxf(m_i[r], v);
      alpha[r] = (m_i[r] == NEG || mn == NEG) ? 0.f : __expf(m_i[r] - mn);
      float p0 = (mn == NEG) ? 0.f : __expf(sc[0][r] - mn);
      float p1 = (mn == NEG) ? 0.f : __expf(sc[1][r] - mn);
      pb[(r + 8 * half) * 32 + lr]      = (__bf16)p0;
      pb[(r + 8 * half) * 32 + 16 + lr] = (__bf16)p1;
      float rs = p0 + p1;
#pragma unroll
      for (int off = 1; off < 16; off <<= 1) rs += __shfl_xor(rs, off, 32);
      l_i[r] = l_i[r] * alpha[r] + rs;
      m_i[r] = mn;
    }
#pragma unroll
    for (int sub = 0; sub < 4; ++sub)
#pragma unroll
      for (int r = 0; r < 8; ++r) o[sub][r] *= alpha[r];

    asm volatile("s_wait_dscnt 0" ::: "memory");
    v16bf pf;
#pragma unroll
    for (int j = 0; j < 8; ++j) {
      pf[j]     = pb[lr * 32 + c0 + j];
      pf[j + 8] = pb[lr * 32 + c0 + 16 + j];
    }
#pragma unroll
    for (int sub = 0; sub < 4; ++sub) {
      v16bf bv;
#pragma unroll
      for (int j = 0; j < 8; ++j) {
        bv[j]     = vbase[(size_t)(kt * 32 + c0 + j)      * TS + sub * 16 + lr];
        bv[j + 8] = vbase[(size_t)(kt * 32 + c0 + 16 + j) * TS + sub * 16 + lr];
      }
      o[sub] = wmma_bf16(pf, bv, o[sub]);
    }
  }
#pragma unroll
  for (int sub = 0; sub < 4; ++sub)
#pragma unroll
    for (int r = 0; r < 8; ++r) {
      int qrow = s0 + r + 8 * half;
      out[((size_t)qrow * BATCH + bb) * HID + h * HDIM + sub * 16 + lr] =
          (__bf16)(o[sub][r] / l_i[r]);
    }
}

// ---------------- router: logits, softmax, top-2 ----------------
__global__ __launch_bounds__(256) void router_kernel(const float* __restrict__ X,
                                                     const float* __restrict__ RW,
                                                     float* __restrict__ probs2,
                                                     int* __restrict__ eidx2) {
  int lane = threadIdx.x & 31;
  int t = blockIdx.x * 8 + (threadIdx.x >> 5);
  const float* xr = X + (size_t)t * HID;
  float lg[NEXP];
#pragma unroll
  for (int e = 0; e < NEXP; ++e) {
    const float* wr = RW + (size_t)e * HID;
    float s = 0.f;
    for (int k = lane; k < HID; k += 32) s += xr[k] * wr[k];
#pragma unroll
    for (int off = 16; off > 0; off >>= 1) s += __shfl_xor(s, off, 32);
    lg[e] = s;
  }
  float mx = lg[0];
#pragma unroll
  for (int e = 1; e < NEXP; ++e) mx = fmaxf(mx, lg[e]);
  float den = 0.f;
#pragma unroll
  for (int e = 0; e < NEXP; ++e) { lg[e] = __expf(lg[e] - mx); den += lg[e]; }
  float inv = 1.f / den;
#pragma unroll
  for (int e = 0; e < NEXP; ++e) lg[e] *= inv;
  int i1 = 0;
#pragma unroll
  for (int e = 1; e < NEXP; ++e) if (lg[e] > lg[i1]) i1 = e;
  int i2 = (i1 == 0) ? 1 : 0;
#pragma unroll
  for (int e = 0; e < NEXP; ++e) if (e != i1 && lg[e] > lg[i2]) i2 = e;
  if (lane == 0) {
    probs2[t * 2] = lg[i1]; probs2[t * 2 + 1] = lg[i2];
    eidx2[t * 2] = i1;      eidx2[t * 2 + 1] = i2;
  }
}

// ---------------- routing setup: 32-padded expert segments ----------------
__global__ __launch_bounds__(256) void route_setup(const int* __restrict__ eidx2,
                                                   int* __restrict__ perm,
                                                   int* __restrict__ tileExpert,
                                                   int* __restrict__ apos) {
  __shared__ int cnt[NEXP], off[NEXP], cur[NEXP];
  int tid = threadIdx.x;
  if (tid < NEXP) { cnt[tid] = 0; cur[tid] = 0; }
  __syncthreads();
  for (int i = tid; i < TOKENS * 2; i += 256) atomicAdd(&cnt[eidx2[i]], 1);
  __syncthreads();
  if (tid == 0) {
    int o = 0;
    for (int e = 0; e < NEXP; ++e) { off[e] = o; o += (cnt[e] + 31) & ~31; }
  }
  __syncthreads();
  for (int ti = tid; ti < MAX_T32; ti += 256) {
    int e = -1, r0 = ti * 32;
    for (int x = 0; x < NEXP; ++x) {
      int pc = (cnt[x] + 31) & ~31;
      if (r0 >= off[x] && r0 < off[x] + pc) e = x;
    }
    tileExpert[ti] = e;
  }
  for (int i = tid; i < MAX_ROWS; i += 256) perm[i] = -1;
  __syncthreads();
  for (int i = tid; i < TOKENS * 2; i += 256) {
    int e = eidx2[i];
    int pos = off[e] + atomicAdd(&cur[e], 1);
    perm[pos] = i >> 1;
    apos[i] = pos;
  }
}

// ---------------- combine: out = h_attn + g0*y[pos0] + g1*y[pos1] ----------------
__global__ __launch_bounds__(256) void combine_kernel(const float* __restrict__ hAttn,
                                                      const float* __restrict__ yPerm,
                                                      const float* __restrict__ probs2,
                                                      const int* __restrict__ apos,
                                                      float* __restrict__ out) {
  int t = blockIdx.x;
  float g0 = probs2[t * 2], g1 = probs2[t * 2 + 1];
  const float* y0 = yPerm + (size_t)apos[t * 2] * HID;
  const float* y1 = yPerm + (size_t)apos[t * 2 + 1] * HID;
  const float* hb = hAttn + (size_t)t * HID;
  float* ob = out + (size_t)t * HID;
#pragma unroll
  for (int j = 0; j < 4; ++j) {
    int i = threadIdx.x + j * 256;
    ob[i] = hb[i] + g0 * y0[i] + g1 * y1[i];
  }
}

// ---------------- host launch ----------------
static size_t align256(size_t x) { return (x + 255) & ~(size_t)255; }

extern "C" void kernel_launch(void* const* d_in, const int* in_sizes, int n_in,
                              void* d_out, int out_size, void* d_ws, size_t ws_size,
                              hipStream_t stream) {
  const float* hidden  = (const float*)d_in[0];
  const float* ln1_w   = (const float*)d_in[1];
  const float* ln1_b   = (const float*)d_in[2];
  const float* ln2_w   = (const float*)d_in[3];
  const float* ln2_b   = (const float*)d_in[4];
  const float* qkv_w   = (const float*)d_in[5];
  const float* proj_w  = (const float*)d_in[6];
  const float* rout_w  = (const float*)d_in[7];
  const float* moe_w1  = (const float*)d_in[8];
  const float* moe_w2  = (const float*)d_in[9];
  float* out = (float*)d_out;

  char* ws = (char*)d_ws;
  size_t off = 0;
  __bf16* ln1_bf  = (__bf16*)(ws + off); off += align256((size_t)TOKENS * HID * 2);
  __bf16* qkv_bf  = (__bf16*)(ws + off); off += align256((size_t)TOKENS * QKVW * 2);
  __bf16* attn_bf = (__bf16*)(ws + off); off += align256((size_t)TOKENS * HID * 2);
  float*  h_attn  = (float*) (ws + off); off += align256((size_t)TOKENS * HID * 4);
  float*  ln2_f   = (float*) (ws + off); off += align256((size_t)TOKENS * HID * 4);
  __bf16* ln2_bf  = (__bf16*)(ws + off); off += align256((size_t)TOKENS * HID * 2);
  float*  probs2  = (float*) (ws + off); off += align256((size_t)TOKENS * 2 * 4);
  int*    eidx2   = (int*)   (ws + off); off += align256((size_t)TOKENS * 2 * 4);
  int*    apos    = (int*)   (ws + off); off += align256((size_t)TOKENS * 2 * 4);
  int*    perm    = (int*)   (ws + off); off += align256((size_t)MAX_ROWS * 4);
  int*    tileExp = (int*)   (ws + off); off += align256((size_t)MAX_T32 * 4);
  __bf16* qkvw_bf = (__bf16*)(ws + off); off += align256((size_t)QKVW * HID * 2);
  __bf16* projw_bf= (__bf16*)(ws + off); off += align256((size_t)HID * HID * 2);
  __bf16* w1t     = (__bf16*)(ws + off); off += align256((size_t)NEXP * HID * FFN * 2);
  __bf16* w2t     = (__bf16*)(ws + off); off += align256((size_t)NEXP * FFN * HID * 2);
  __bf16* h1Perm  = (__bf16*)(ws + off); off += align256((size_t)MAX_ROWS * FFN * 2);
  float*  yPerm   = (float*) (ws + off); off += align256((size_t)MAX_ROWS * HID * 4);
  (void)ws_size; (void)in_sizes; (void)n_in; (void)out_size;

  // 0) one-shot weight preprocessing (bf16 convert; MoE weights transposed to [E][N][K])
  cvt_bf16_kernel<<<(QKVW * HID) / 1024, 256, 0, stream>>>(qkv_w, qkvw_bf, (size_t)QKVW * HID);
  cvt_bf16_kernel<<<(HID * HID) / 1024, 256, 0, stream>>>(proj_w, projw_bf, (size_t)HID * HID);
  transpose_w_kernel<<<dim3(HID / 64, FFN / 64, NEXP), 256, 0, stream>>>(moe_w1, w1t, HID, FFN);
  transpose_w_kernel<<<dim3(FFN / 64, HID / 64, NEXP), 256, 0, stream>>>(moe_w2, w2t, FFN, HID);

  // 1) LN1 -> bf16
  ln_kernel<<<TOKENS, 256, 0, stream>>>(hidden, ln1_w, ln1_b, nullptr, ln1_bf);
  // 2) QKV = ln1 x qkv_w^T  -> bf16
  gemm_bf<0, 0, 1, 0><<<dim3(TOKENS / 256, QKVW / 64), 256, 0, stream>>>(
      ln1_bf, qkvw_bf, nullptr, nullptr, nullptr, qkv_bf, TOKENS, QKVW, HID, 0);
  // 3) causal GQA flash attention -> bf16
  attn_kernel<<<(SEQ / 16) * BATCH * NHEADS / 8, 256, 0, stream>>>(qkv_bf, attn_bf);
  // 4) h_attn = hidden + attn x proj_w^T  -> f32
  gemm_bf<0, 0, 0, 1><<<dim3(TOKENS / 256, HID / 64), 256, 0, stream>>>(
      attn_bf, projw_bf, nullptr, nullptr, hidden, h_attn, TOKENS, HID, HID, 0);
  // 5) LN2 -> f32 (router) + bf16 (MoE)
  ln_kernel<<<TOKENS, 256, 0, stream>>>(h_attn, ln2_w, ln2_b, ln2_f, ln2_bf);
  // 6) router + top-2
  router_kernel<<<TOKENS / 8, 256, 0, stream>>>(ln2_f, rout_w, probs2, eidx2);
  // 7) padded expert segments (device-side, static grid; graph-safe)
  route_setup<<<1, 256, 0, stream>>>(eidx2, perm, tileExp, apos);
  // 8) sparse top-2 MoE GEMMs (gathered tokens, per-expert transposed bf16 weights)
  gemm_bf<1, 1, 1, 0><<<dim3(MAX_T32 / 8, FFN / 64), 256, 0, stream>>>(
      ln2_bf, w1t, perm, tileExp, nullptr, h1Perm, MAX_ROWS, FFN, HID, (long)HID * FFN);
  gemm_bf<0, 0, 0, 0><<<dim3(MAX_T32 / 8, HID / 64), 256, 0, stream>>>(
      h1Perm, w2t, nullptr, tileExp, nullptr, yPerm, MAX_ROWS, HID, FFN, (long)FFN * HID);
  // 9) deterministic gather-combine
  combine_kernel<<<TOKENS, 256, 0, stream>>>(h_attn, yPerm, probs2, apos, out);
}
